// ResNet_37546604102380
// MI455X (gfx1250) — matrix operand order repaired and checked by reading, not compiled, not run on hardware
//
#include <hip/hip_runtime.h>

// ---------------------------------------------------------------------------
// CDNA5 (gfx1250, wave32) WMMA bf16 implementation of the point-cloud ResNet.
// All heavy math is V_WMMA_F32_16X16X32_BF16 GEMMs with fused BN/ReLU/residual.
// Per-wave tile 32x64 -> 8 WMMA per K-step; division-free staging loops.
// ---------------------------------------------------------------------------

typedef __attribute__((ext_vector_type(16))) __bf16 v16bf;
typedef __attribute__((ext_vector_type(8)))  __bf16 v8bf;
typedef __attribute__((ext_vector_type(8)))  float  v8f;

#define NPTS 2048
#define NBATCH 2
#define CMAX 2048

// GEMM tiling
#define BM 128    // output rows per block
#define BN_T 128  // output cols per block
#define BK 32     // K per step (one WMMA K)
#define LDA 48    // LDS row stride (bf16 elems) for A tile (128 x 32)
#define LDB 48    // LDS row stride (bf16 elems) for B tile (128 x 32, n-major)

__device__ __forceinline__ unsigned short f2bf(float f) {
  union { float f; unsigned u; } v; v.f = f;
  unsigned u = v.u;
  u += 0x7fffu + ((u >> 16) & 1u);   // round-to-nearest-even
  return (unsigned short)(u >> 16);
}

// ---------------------------------------------------------------------------
// Fused GEMM:  Out[b] (O x Npts) = W (O x K, K-major) * X[b] (C x Npts)
//   mode 0: plain 1x1 conv, K = C
//   mode 1: wconv, K = 9*C (W pre-repacked K-major), B built on the fly:
//           B[j*C+c, n] = X[c, n + (j-4)*dil] * wtab[j, n]
// Epilogue: y = acc*scale[o] + shift[o] (+ residual) (relu)
// ---------------------------------------------------------------------------
__global__ __launch_bounds__(256) void gemm_wmma_fused(
    const float* __restrict__ W, const float* __restrict__ X,
    float* __restrict__ Out, const float* __restrict__ Res,
    const float* __restrict__ bg, const float* __restrict__ bb,
    const float* __restrict__ bm, const float* __restrict__ bv,
    const float* __restrict__ wtab,   // (NBATCH, 9, Npts) for mode 1
    int O, int C, int K, int mode, int dil, int relu, int hasRes)
{
  const int b = blockIdx.z;
  const float* Xb = X + (size_t)b * C * NPTS;
  float* Outb = Out + (size_t)b * O * NPTS;
  const float* Resb = Res ? (Res + (size_t)b * O * NPTS) : nullptr;
  const float* wtb = wtab ? (wtab + (size_t)b * 9 * NPTS) : nullptr;

  const int m0 = blockIdx.y * BM;
  const int n0 = blockIdx.x * BN_T;

  __shared__ alignas(32) unsigned short sA[BM * LDA];
  __shared__ alignas(32) unsigned short sB[BN_T * LDB];

  const int t = threadIdx.x;
  const int wave = t >> 5;           // 8 waves (wave32)
  const int lane = t & 31;
  const int wm = wave & 3;           // 4 M-groups of 32 rows
  const int wn = wave >> 2;          // 2 N-halves of 64 cols

  v8f acc[2][4] = {};                // 2 M-strips x 4 N-tiles of 16x16 f32

  // per-K-step uniforms for mode 1 (no divisions anywhere)
  int jj = 0;                        // tap index kg/C
  int c0 = 0;                        // kg - jj*C for this step
  int off = -4 * dil;                // (jj-4)*dil

  for (int k0 = 0; k0 < K; k0 += BK) {
    // ---- stage A tile (128 x 32): 4 float4 loads/thread, K-major ----
#pragma unroll
    for (int it = 0; it < 4; ++it) {
      int e  = t + 256 * it;
      int k4 = (e & 7) << 2;         // 0,4,...,28
      int oo = e >> 3;               // 0..127
      int og = m0 + oo;
      float4 v = make_float4(0.f, 0.f, 0.f, 0.f);
      if (og < O) v = *(const float4*)&W[og * K + k0 + k4];
      union { unsigned short us[4]; uint2 u2; } p;
      p.us[0] = f2bf(v.x); p.us[1] = f2bf(v.y);
      p.us[2] = f2bf(v.z); p.us[3] = f2bf(v.w);
      *(uint2*)&sA[oo * LDA + k4] = p.u2;
    }
    // ---- stage B tile (32 x 128), stored n-major in LDS ----
    if (mode == 0) {
#pragma unroll
      for (int it = 0; it < 4; ++it) {
        int e  = t + 256 * it;
        int n4 = (e & 31) << 2;      // 0,4,...,124
        int kk = e >> 5;             // 0..31
        float4 v = *(const float4*)&Xb[(k0 + kk) * NPTS + n0 + n4];
        sB[(n4 + 0) * LDB + kk] = f2bf(v.x);
        sB[(n4 + 1) * LDB + kk] = f2bf(v.y);
        sB[(n4 + 2) * LDB + kk] = f2bf(v.z);
        sB[(n4 + 3) * LDB + kk] = f2bf(v.w);
      }
      if (k0 + BK < K)   // global_prefetch_b8 of next K slab
        __builtin_prefetch(&Xb[(k0 + BK + (t >> 3)) * NPTS + n0 + ((t & 7) << 4)]);
    } else {
      const float* xrow = Xb + c0 * NPTS;    // channel base for this step
      const float* wtrow = wtb + jj * NPTS;  // tap weights for this step
#pragma unroll
      for (int it = 0; it < 16; ++it) {
        int e  = t + 256 * it;
        int nn = e & 127;
        int kk = e >> 7;             // 0..31 (same tap jj for whole step)
        int ng = n0 + nn;
        int src = ng + off;
        float val = 0.f;
        if (src >= 0 && src < NPTS) val = xrow[kk * NPTS + src] * wtrow[ng];
        sB[nn * LDB + kk] = f2bf(val);
      }
    }
    __syncthreads();

    // ---- fragments & 8 WMMA ----
    // A (16-bit 16x32): lanes 0-15 -> K 0..15, lanes 16-31 -> K 16..31 (contiguous)
    // B (32x16): lanes 0-15 -> K 0..15, lanes 16-31 -> K 16..31 (contiguous)
    const int l15   = lane & 15;
    const int koff  = (lane >> 4) * 8;
    const int koff2 = (lane >> 4) * 16;
    v16bf bfrag[4];
#pragma unroll
    for (int sn = 0; sn < 4; ++sn)
      bfrag[sn] = *(const v16bf*)&sB[(wn * 64 + sn * 16 + l15) * LDB + koff2];
#pragma unroll
    for (int s = 0; s < 2; ++s) {
      const int arow = wm * 32 + s * 16 + l15;
      v8bf alo = *(const v8bf*)&sA[arow * LDA + koff2];
      v8bf ahi = *(const v8bf*)&sA[arow * LDA + koff2 + 8];
      v16bf afrag = __builtin_shufflevector(alo, ahi,
          0, 1, 2, 3, 4, 5, 6, 7, 8, 9, 10, 11, 12, 13, 14, 15);
#pragma unroll
      for (int sn = 0; sn < 4; ++sn)
        acc[s][sn] = __builtin_amdgcn_wmma_f32_16x16x32_bf16(
            false, afrag, false, bfrag[sn], (short)0, acc[s][sn], false, false);
    }
    __syncthreads();

    c0 += BK;
    if (c0 >= C) { c0 = 0; ++jj; off += dil; }
  }

  // ---- epilogue: BN fold + residual + ReLU ----
  // C/D layout: VGPR r, lanes 0-15: M=r, N=lane; lanes 16-31: M=r+8, N=lane-16
  const int l15 = lane & 15;
  const int rowadd = (lane >> 4) * 8;
  const int colbase = n0 + wn * 64;
#pragma unroll
  for (int s = 0; s < 2; ++s) {
    const int rbase = m0 + wm * 32 + s * 16 + rowadd;
    if (rbase >= O) continue;        // wave-uniform (O is a multiple of 64)
#pragma unroll
    for (int r = 0; r < 8; ++r) {
      const int row = rbase + r;
      const float sc = bg[row] * rsqrtf(bv[row] + 1e-5f);
      const float sh = bb[row] - bm[row] * sc;
#pragma unroll
      for (int sn = 0; sn < 4; ++sn) {
        const int col = colbase + sn * 16 + l15;
        float o = acc[s][sn][r] * sc + sh;
        if (hasRes) o += Resb[row * NPTS + col];
        if (relu) o = fmaxf(o, 0.f);
        Outb[row * NPTS + col] = o;
      }
    }
  }
}

// ---------------------------------------------------------------------------
// Repack wconv weights (O,C,9) -> K-major (O, 9*C): Wt[o][j*C+c] = W[o][c][j]
// ---------------------------------------------------------------------------
__global__ void repack_w2(const float* __restrict__ W, float* __restrict__ Wt,
                          int O, int C)
{
  int idx = blockIdx.x * blockDim.x + threadIdx.x;  // over O*C*9, read-coalesced
  int j = idx % 9;
  int c = (idx / 9) % C;
  int o = idx / (9 * C);
  Wt[o * 9 * C + j * C + c] = W[idx];
}

// ---------------------------------------------------------------------------
// Gather x and coords into permuted order (done ONCE for the whole network).
// ---------------------------------------------------------------------------
__global__ void gather_kernel(const float* __restrict__ x,
                              const float* __restrict__ coords,
                              const int* __restrict__ indices,
                              float* __restrict__ xp, float* __restrict__ cp)
{
  int idx = blockIdx.x * blockDim.x + threadIdx.x;   // over NBATCH*3*NPTS
  int n = idx % NPTS;
  int c = (idx / NPTS) % 3;
  int b = idx / (3 * NPTS);
  int src = indices[b * NPTS + n];
  int o = (b * 3 + c) * NPTS;
  xp[o + n] = x[o + src];
  cp[o + n] = coords[o + src];
}

// ---------------------------------------------------------------------------
// Precompute the 7 distinct (sigma, dilation) Gaussian tap tables.
// ---------------------------------------------------------------------------
__global__ void wtab_kernel(const float* __restrict__ cp, float* __restrict__ wtab)
{
  int idx = blockIdx.x * blockDim.x + threadIdx.x;   // over NBATCH*9*NPTS
  int cfg = blockIdx.y;                              // 0..6
  int n = idx % NPTS;
  int j = (idx / NPTS) % 9;
  int b = idx / (9 * NPTS);
  float sig; int dil;
  switch (cfg) {
    case 0: sig = 1.f;  dil = 1; break;
    case 1: sig = 4.f;  dil = 1; break;
    case 2: sig = 1.f;  dil = 2; break;
    case 3: sig = 8.f;  dil = 2; break;
    case 4: sig = 1.f;  dil = 4; break;
    case 5: sig = 16.f; dil = 4; break;
    default: sig = 1.f; dil = 8; break;
  }
  int src = n + (j - 4) * dil;
  bool in = (src >= 0 && src < NPTS);
  float d2 = 0.f;
  const float* cb = cp + b * 3 * NPTS;
#pragma unroll
  for (int c = 0; c < 3; ++c) {
    float a  = cb[c * NPTS + n];
    float bc = in ? cb[c * NPTS + src] : 0.f;
    float d = a - bc;
    d2 += d * d;
  }
  float inv2s2 = 1.f / (2.f * sig * sig);
  wtab[((cfg * NBATCH + b) * 9 + j) * NPTS + n] = __expf(-d2 * inv2s2);
}

// ---------------------------------------------------------------------------
// Stem wconv (C=3, trivial FLOPs) + BN + ReLU, scalar.
// ---------------------------------------------------------------------------
__global__ void stem_kernel(const float* __restrict__ xp,
                            const float* __restrict__ Wst,   // (64,3,9)
                            const float* __restrict__ wt0,   // cfg0 table
                            const float* __restrict__ g, const float* __restrict__ bb,
                            const float* __restrict__ m, const float* __restrict__ v,
                            float* __restrict__ out)
{
  int idx = blockIdx.x * blockDim.x + threadIdx.x;   // over NBATCH*64*NPTS
  int n = idx % NPTS;
  int o = (idx / NPTS) % 64;
  int b = idx / (64 * NPTS);
  const float* xb  = xp  + b * 3 * NPTS;
  const float* wtb = wt0 + b * 9 * NPTS;
  float acc = 0.f;
#pragma unroll
  for (int j = 0; j < 9; ++j) {
    int src = n + (j - 4);
    if (src < 0 || src >= NPTS) continue;
    float w = wtb[j * NPTS + n];
#pragma unroll
    for (int c = 0; c < 3; ++c)
      acc += Wst[(o * 3 + c) * 9 + j] * xb[c * NPTS + src] * w;
  }
  float s  = g[o] * rsqrtf(v[o] + 1e-5f);
  float sh = bb[o] - m[o] * s;
  out[(b * 64 + o) * NPTS + n] = fmaxf(acc * s + sh, 0.f);
}

// ---------------------------------------------------------------------------
// Final scatter back to original point order.
// ---------------------------------------------------------------------------
__global__ void scatter_kernel(const float* __restrict__ h,
                               const int* __restrict__ reind,
                               float* __restrict__ out)
{
  int idx = blockIdx.x * blockDim.x + threadIdx.x;   // over NBATCH*CMAX*NPTS
  int n = idx % NPTS;
  int c = (idx / NPTS) % CMAX;
  int b = idx / (CMAX * NPTS);
  int r = reind[b * NPTS + n];
  size_t base = (size_t)(b * CMAX + c) * NPTS;
  out[base + n] = h[base + r];
}

// ---------------------------------------------------------------------------
// Host orchestration
// ---------------------------------------------------------------------------
static int cfg_of(float sig, int dil) {
  if (sig == 1.f  && dil == 1) return 0;
  if (sig == 4.f  && dil == 1) return 1;
  if (sig == 1.f  && dil == 2) return 2;
  if (sig == 8.f  && dil == 2) return 3;
  if (sig == 1.f  && dil == 4) return 4;
  if (sig == 16.f && dil == 4) return 5;
  return 6;                                   // (1.0, 8)
}

extern "C" void kernel_launch(void* const* d_in, const int* in_sizes, int n_in,
                              void* d_out, int out_size, void* d_ws, size_t ws_size,
                              hipStream_t stream)
{
  const float* x         = (const float*)d_in[0];
  const float* coords    = (const float*)d_in[1];
  const int*   indices   = (const int*)d_in[2];
  const int*   reindices = (const int*)d_in[3];
  int p = 4;
  const float* stem_w = (const float*)d_in[p++];
  const float* sbn[4];
  for (int i = 0; i < 4; ++i) sbn[i] = (const float*)d_in[p++];

  struct Cfg { int planes, nb, dfirst, drest; float sfirst; };
  const Cfg LYR[4] = {{64,3,1,1,4.f},{128,4,1,2,4.f},{256,6,2,4,8.f},{512,3,4,8,16.f}};

  struct Blk {
    const float *w1, *w2, *w3, *dw;
    const float *b1[4], *b2[4], *b3[4], *db[4];
    int width, inpl, outc, cfg, dil, hasDw;
  };
  Blk blk[16];
  int nblk = 0, inpl = 64;
  for (int si = 0; si < 4; ++si) {
    for (int bi = 0; bi < LYR[si].nb; ++bi) {
      Blk& B = blk[nblk++];
      B.width = LYR[si].planes; B.inpl = inpl; B.outc = LYR[si].planes * 4;
      B.w1 = (const float*)d_in[p++];
      for (int i = 0; i < 4; ++i) B.b1[i] = (const float*)d_in[p++];
      B.w2 = (const float*)d_in[p++];
      for (int i = 0; i < 4; ++i) B.b2[i] = (const float*)d_in[p++];
      B.w3 = (const float*)d_in[p++];
      for (int i = 0; i < 4; ++i) B.b3[i] = (const float*)d_in[p++];
      B.hasDw = (bi == 0 && inpl != B.outc);
      B.dw = nullptr;
      if (B.hasDw) {
        B.dw = (const float*)d_in[p++];
        for (int i = 0; i < 4; ++i) B.db[i] = (const float*)d_in[p++];
      }
      B.dil = (bi == 0) ? LYR[si].dfirst : LYR[si].drest;
      B.cfg = cfg_of((bi == 0) ? LYR[si].sfirst : 1.f, B.dil);
      inpl = B.outc;
    }
  }

  // workspace carve-up (floats)
  float* ws = (float*)d_ws;
  const size_t actsz = (size_t)NBATCH * CMAX * NPTS;
  float* hA = ws;
  float* hB = hA + actsz;
  float* hC = hB + actsz;
  float* hD = hC + actsz;
  float* xp = hD + actsz;
  float* cp = xp + (size_t)NBATCH * 3 * NPTS;
  float* wtab = cp + (size_t)NBATCH * 3 * NPTS;          // 7 * NBATCH * 9 * NPTS
  float* wrep = wtab + (size_t)7 * NBATCH * 9 * NPTS;    // <= 512 * 9 * 512 floats

  gather_kernel<<<(NBATCH * 3 * NPTS) / 256, 256, 0, stream>>>(x, coords, indices, xp, cp);
  dim3 wg((NBATCH * 9 * NPTS) / 256, 7);
  wtab_kernel<<<wg, 256, 0, stream>>>(cp, wtab);
  stem_kernel<<<(NBATCH * 64 * NPTS) / 256, 256, 0, stream>>>(
      xp, stem_w, wtab /* cfg 0 */, sbn[0], sbn[1], sbn[2], sbn[3], hA);

  const size_t cfgstride = (size_t)NBATCH * 9 * NPTS;
  for (int i = 0; i < nblk; ++i) {
    Blk& B = blk[i];
    const float* wt = wtab + (size_t)B.cfg * cfgstride;

    // o1 = relu(bn1(W1 @ h))
    {
      dim3 g(NPTS / BN_T, (B.width + BM - 1) / BM, NBATCH);
      gemm_wmma_fused<<<g, 256, 0, stream>>>(
          B.w1, hA, hB, nullptr, B.b1[0], B.b1[1], B.b1[2], B.b1[3],
          nullptr, B.width, B.inpl, B.inpl, 0, 0, 1, 0);
    }
    // o2 = relu(bn2(wconv(o1)))  -- repack w2 K-major, then K = 9*width GEMM
    {
      int nel = B.width * B.width * 9;
      repack_w2<<<nel / 256, 256, 0, stream>>>(B.w2, wrep, B.width, B.width);
      dim3 g(NPTS / BN_T, (B.width + BM - 1) / BM, NBATCH);
      gemm_wmma_fused<<<g, 256, 0, stream>>>(
          wrep, hB, hC, nullptr, B.b2[0], B.b2[1], B.b2[2], B.b2[3],
          wt, B.width, B.width, 9 * B.width, 1, B.dil, 1, 0);
    }
    // identity path (optional downsample)
    const float* resid = hA;
    if (B.hasDw) {
      dim3 g(NPTS / BN_T, (B.outc + BM - 1) / BM, NBATCH);
      gemm_wmma_fused<<<g, 256, 0, stream>>>(
          B.dw, hA, hB, nullptr, B.db[0], B.db[1], B.db[2], B.db[3],
          nullptr, B.outc, B.inpl, B.inpl, 0, 0, 0, 0);
      resid = hB;
    }
    // h = relu(bn3(W3 @ o2) + ident)
    {
      dim3 g(NPTS / BN_T, (B.outc + BM - 1) / BM, NBATCH);
      gemm_wmma_fused<<<g, 256, 0, stream>>>(
          B.w3, hC, hD, resid, B.b3[0], B.b3[1], B.b3[2], B.b3[3],
          nullptr, B.outc, B.width, B.width, 0, 0, 1, 1);
    }
    float* tmp = hA; hA = hD; hD = tmp;
  }

  scatter_kernel<<<(int)(actsz / 256), 256, 0, stream>>>(hA, reindices, (float*)d_out);
}
